// TULayer_47072841564817
// MI455X (gfx1250) — matrix operand order repaired and checked by reading, not compiled, not run on hardware
//
#include <hip/hip_runtime.h>
#include <stdint.h>

// ---------------------------------------------------------------------------
// Types for CDNA5 WMMA (wave32): v_wmma_f32_16x16x32_bf16
// ---------------------------------------------------------------------------
typedef __attribute__((ext_vector_type(16))) __bf16    v16bf;
typedef __attribute__((ext_vector_type(8)))  float     v8f;
typedef __attribute__((ext_vector_type(4)))  unsigned  u32x4;

union Frag {
    v16bf v;
    u32x4 q[2];
    unsigned u[8];
};

__device__ __forceinline__ unsigned short f2bf(float f) {
    unsigned u = __float_as_uint(f);
    unsigned r = u + 0x7FFFu + ((u >> 16) & 1u);   // round-to-nearest-even
    return (unsigned short)(r >> 16);
}
__device__ __forceinline__ unsigned pack2bf(float lo, float hi) {
    return (unsigned)f2bf(lo) | ((unsigned)f2bf(hi) << 16);
}

__device__ __forceinline__ void wait_async0() {
#if __has_builtin(__builtin_amdgcn_s_wait_asynccnt)
    __builtin_amdgcn_s_wait_asynccnt(0);
#else
    asm volatile("s_wait_asynccnt 0x0" ::: "memory");
#endif
}

#define NSLOT 64   // striping slots for BN-statistics atomics

// ---------------------------------------------------------------------------
// Repack weights (fp32 [taps][K][C]) into WMMA B-fragment order:
//   P[tap][C/16][K/32][lane(32)][16 bf16]
// B layout (16-bit, 32x16): lane<16 -> N=lane, K=0..7,16..23 ; lane>=16 -> K=8..15,24..31.
// Within a lane element e maps to K = (e&7) + 8*half + 16*(e>>3).
// ---------------------------------------------------------------------------
__global__ void pack_w(const float* __restrict__ W, __bf16* __restrict__ P,
                       int K, int C, int total)
{
    int gid = blockIdx.x * 256 + threadIdx.x;
    if (gid >= total) return;
    int nk = K >> 5;             // K-steps of 32
    int nc = C >> 4;             // N-subtiles of 16
    int e    = gid & 15;
    int lane = (gid >> 4) & 31;
    int rest = gid >> 9;
    int s   = rest % nk; rest /= nk;
    int c16 = rest % nc;
    int tp  = rest / nc;
    int n  = lane & 15;
    int kh = lane >> 4;
    int Kidx = s * 32 + kh * 8 + (e >> 3) * 16 + (e & 7);
    int Cidx = c16 * 16 + n;
    unsigned short b = f2bf(W[((size_t)tp * K + Kidx) * C + Cidx]);
    union { unsigned short u; __bf16 b; } cv; cv.u = b;
    P[gid] = cv.b;
}

// ---------------------------------------------------------------------------
// Stage 1: up_raw[N][64] = feats_a[a_idx[i]] @ W_t[k_id[i]]   (bf16 WMMA)
// 4 waves/block (one per 16-col subtile).  Per-k row masking is per-lane
// (A layout holds one row per lane).  Accumulates BN1 sum / sumsq.
// ---------------------------------------------------------------------------
__global__ __launch_bounds__(128)
void stage1_wmma(const float* __restrict__ feats_a,
                 const int*   __restrict__ a_idx,
                 const int*   __restrict__ k_id,
                 const __bf16* __restrict__ WtP,
                 float* __restrict__ up_raw,
                 float* __restrict__ slots,
                 int N, int ntiles, int tiles_per_block)
{
    __shared__ alignas(16) __bf16 sA[16 * 128];
    __shared__ float sS[64], sQ[64];

    const int t    = threadIdx.x;
    const int lane = t & 31;
    const int w    = t >> 5;          // 0..3: column subtile
    const int nloc = lane & 15;
    const int lh   = lane >> 4;

    if (t < 64) { sS[t] = 0.f; sQ[t] = 0.f; }

    const int tile0 = blockIdx.x * tiles_per_block;
    for (int tt = 0; tt < tiles_per_block; ++tt) {
        int tile = tile0 + tt;
        if (tile >= ntiles) break;
        int row0 = tile * 16;

        __syncthreads();   // protect LDS reuse
        // ---- stage A tile: 16 rows x 128 ch, fp32 -> bf16 ----
        {
            int r  = t >> 3;          // 0..15
            int c0 = (t & 7) * 16;    // 0,16,...,112
            int i  = row0 + r;
            unsigned wb[8];
            if (i < N) {
                const float* src = feats_a + (size_t)a_idx[i] * 128 + c0;
                #pragma unroll
                for (int j = 0; j < 8; ++j) wb[j] = pack2bf(src[2*j], src[2*j+1]);
            } else {
                #pragma unroll
                for (int j = 0; j < 8; ++j) wb[j] = 0u;
            }
            u32x4* dst = (u32x4*)&sA[r * 128 + c0];
            dst[0] = (u32x4){wb[0], wb[1], wb[2], wb[3]};
            dst[1] = (u32x4){wb[4], wb[5], wb[6], wb[7]};
        }
        __syncthreads();

        // per-lane row id for masking (A layout: lane holds row M = lane&15)
        int gi  = row0 + nloc;
        int kid = (gi < N) ? k_id[gi] : -1;

        // A fragments: 16-bit A 16x32 layout -> two contiguous b128 LDS reads
        Frag af[4];
        #pragma unroll
        for (int s = 0; s < 4; ++s) {
            af[s].q[0] = *(const u32x4*)&sA[nloc * 128 + s * 32 + lh * 8];
            af[s].q[1] = *(const u32x4*)&sA[nloc * 128 + s * 32 + 16 + lh * 8];
        }

        v8f acc = {};
        #pragma unroll
        for (int k = 0; k < 8; ++k) {
            unsigned msk = (kid == k) ? 0xFFFFFFFFu : 0u;
            #pragma unroll
            for (int s = 0; s < 4; ++s) {
                Frag am;
                #pragma unroll
                for (int j = 0; j < 8; ++j) am.u[j] = af[s].u[j] & msk;
                Frag b;
                const u32x4* bp =
                    (const u32x4*)(WtP + (size_t)((k * 4 + w) * 4 + s) * 512 + lane * 16);
                b.q[0] = bp[0]; b.q[1] = bp[1];
                acc = __builtin_amdgcn_wmma_f32_16x16x32_bf16(
                          false, am.v, false, b.v, (short)0, acc, false, false);
            }
        }

        // epilogue: D layout lane -> (N=lane&15, Mhalf=lane>>4), acc[j] -> M=lh*8+j
        int ch = w * 16 + nloc;
        float ss = 0.f, sq = 0.f;
        #pragma unroll
        for (int j = 0; j < 8; ++j) {
            int oi = row0 + lh * 8 + j;
            float vv = acc[j];
            if (oi < N) {
                up_raw[(size_t)oi * 64 + ch] = vv;
                ss += vv; sq += vv * vv;
            }
        }
        atomicAdd(&sS[ch], ss);
        atomicAdd(&sQ[ch], sq);
    }

    __syncthreads();
    if (t < 64) {
        int slot = blockIdx.x & (NSLOT - 1);
        atomicAdd(&slots[t * NSLOT + slot], sS[t]);
        atomicAdd(&slots[64 * NSLOT + t * NSLOT + slot], sQ[t]);
    }
}

// ---------------------------------------------------------------------------
// BN finalize: reduce striped sums -> per-channel scale/shift
// ---------------------------------------------------------------------------
__global__ void bn_finalize(const float* __restrict__ slots, int C, float invN,
                            const float* __restrict__ gamma,
                            const float* __restrict__ beta,
                            float* __restrict__ scale, float* __restrict__ shift)
{
    int c = threadIdx.x;
    if (c >= C) return;
    float S = 0.f, Q = 0.f;
    for (int s = 0; s < NSLOT; ++s) {
        S += slots[c * NSLOT + s];
        Q += slots[C * NSLOT + c * NSLOT + s];
    }
    float mu  = S * invN;
    float var = Q * invN - mu * mu;
    float inv = rsqrtf(var + 1e-5f);
    float sc  = gamma[c] * inv;
    scale[c] = sc;
    shift[c] = beta[c] - mu * sc;
}

// ---------------------------------------------------------------------------
// Build cat (N x 128, bf16): [relu(BN1(up)) | feats_b]  -- L2-resident table
// ---------------------------------------------------------------------------
__global__ void build_cat(const float* __restrict__ up_raw,
                          const float* __restrict__ feats_b,
                          const float* __restrict__ scale,
                          const float* __restrict__ shift,
                          __bf16* __restrict__ cat, int N)
{
    long long gid = (long long)blockIdx.x * 256 + threadIdx.x;
    if (gid >= (long long)N * 16) return;
    int g = (int)(gid & 15);
    long long i = gid >> 4;
    unsigned wb[4];
    if (g < 8) {
        int c0 = g * 8;
        const float* src = up_raw + i * 64 + c0;
        #pragma unroll
        for (int j = 0; j < 4; ++j) {
            float a = fmaxf(src[2*j]   * scale[c0+2*j]   + shift[c0+2*j],   0.f);
            float b = fmaxf(src[2*j+1] * scale[c0+2*j+1] + shift[c0+2*j+1], 0.f);
            wb[j] = pack2bf(a, b);
        }
        *(u32x4*)(cat + i * 128 + c0) = (u32x4){wb[0], wb[1], wb[2], wb[3]};
    } else {
        int c0 = (g - 8) * 8;
        const float* src = feats_b + i * 64 + c0;
        #pragma unroll
        for (int j = 0; j < 4; ++j) wb[j] = pack2bf(src[2*j], src[2*j+1]);
        *(u32x4*)(cat + i * 128 + 64 + c0) = (u32x4){wb[0], wb[1], wb[2], wb[3]};
    }
}

// ---------------------------------------------------------------------------
// Async mask-aware gather of one 16-row x 256B tap tile into LDS.
// Hit chunks use GLOBAL_LOAD_ASYNC_TO_LDS_B128 (ASYNCcnt, no VGPR round-trip);
// miss chunks are zero-filled with ds_store_b128 (DScnt).
// ---------------------------------------------------------------------------
__device__ __forceinline__ void stage_tap_async(const __bf16* __restrict__ cat,
                                                const int*    __restrict__ nbr_idx,
                                                const float*  __restrict__ nbr_mask,
                                                __bf16* buf, int k, int row0,
                                                int N, int t)
{
    #pragma unroll
    for (int cbase = 0; cbase < 384; cbase += 192) {
        int c = cbase + t;
        if (c < 256) {
            int r = c >> 4;
            int p = c & 15;
            int i = row0 + r;
            __bf16* dst = &buf[r * 128 + p * 8];
            bool hit = false;
            const __bf16* src = cat;
            if (i < N) {
                float mk = nbr_mask[(size_t)k * N + i];
                if (mk != 0.f) {
                    hit = true;
                    src = cat + (size_t)nbr_idx[(size_t)k * N + i] * 128 + p * 8;
                }
            }
            if (hit) {
                // wave-relative LDS byte offset = low 32 bits of generic ptr
                unsigned lds = (unsigned)(size_t)(void*)dst;
                asm volatile("global_load_async_to_lds_b128 %0, %1, off"
                             :: "v"(lds), "v"(src) : "memory");
            } else {
                *(u32x4*)dst = (u32x4){0u, 0u, 0u, 0u};
            }
        }
    }
}

// ---------------------------------------------------------------------------
// Stage 3: out_raw[N][96] = sum_k (mask_k * cat[nbr_idx_k]) @ W_out[k]
// 6 waves/block (COUT=96).  Software-pipelined: tap k+1 is async-gathered into
// the other LDS buffer while tap k's 4 WMMAs execute; one barrier per tap.
// ---------------------------------------------------------------------------
__global__ __launch_bounds__(192)
void stage3_wmma(const __bf16* __restrict__ cat,
                 const int*    __restrict__ nbr_idx,
                 const float*  __restrict__ nbr_mask,
                 const __bf16* __restrict__ WoP,
                 float* __restrict__ out_raw,
                 float* __restrict__ slots,
                 int N, int ntiles, int tiles_per_block)
{
    __shared__ alignas(16) __bf16 sA[2][16 * 128];
    __shared__ float sS[96], sQ[96];

    const int t    = threadIdx.x;
    const int lane = t & 31;
    const int w    = t >> 5;      // 0..5
    const int nloc = lane & 15;
    const int lh   = lane >> 4;

    if (t < 96) { sS[t] = 0.f; sQ[t] = 0.f; }

    const int tile0 = blockIdx.x * tiles_per_block;
    for (int tt = 0; tt < tiles_per_block; ++tt) {
        int tile = tile0 + tt;
        if (tile >= ntiles) break;
        int row0 = tile * 16;

        // prologue: gather tap 0 into buf0
        stage_tap_async(cat, nbr_idx, nbr_mask, sA[0], 0, row0, N, t);
        wait_async0();
        __syncthreads();

        v8f acc = {};
        for (int k = 0; k < 27; ++k) {
            // overlap: gather tap k+1 while computing tap k
            if (k < 26)
                stage_tap_async(cat, nbr_idx, nbr_mask, sA[(k + 1) & 1],
                                k + 1, row0, N, t);

            const __bf16* buf = sA[k & 1];
            #pragma unroll
            for (int s = 0; s < 4; ++s) {
                Frag a;
                a.q[0] = *(const u32x4*)&buf[nloc * 128 + s * 32 + lh * 8];
                a.q[1] = *(const u32x4*)&buf[nloc * 128 + s * 32 + 16 + lh * 8];
                Frag b;
                const u32x4* bp =
                    (const u32x4*)(WoP + (size_t)((k * 6 + w) * 4 + s) * 512 + lane * 16);
                b.q[0] = bp[0]; b.q[1] = bp[1];
                acc = __builtin_amdgcn_wmma_f32_16x16x32_bf16(
                          false, a.v, false, b.v, (short)0, acc, false, false);
            }

            wait_async0();      // tap k+1 LDS writes complete (had wmma time to land)
            __syncthreads();    // visibility + everyone done reading buf[k&1]
        }

        int ch = w * 16 + nloc;   // < 96
        float ss = 0.f, sq = 0.f;
        #pragma unroll
        for (int j = 0; j < 8; ++j) {
            int oi = row0 + lh * 8 + j;
            float vv = acc[j];
            if (oi < N) {
                out_raw[(size_t)oi * 96 + ch] = vv;
                ss += vv; sq += vv * vv;
            }
        }
        atomicAdd(&sS[ch], ss);
        atomicAdd(&sQ[ch], sq);
    }

    __syncthreads();
    if (t < 96) {
        int slot = blockIdx.x & (NSLOT - 1);
        atomicAdd(&slots[t * NSLOT + slot], sS[t]);
        atomicAdd(&slots[96 * NSLOT + t * NSLOT + slot], sQ[t]);
    }
}

// ---------------------------------------------------------------------------
// Final BN2 affine + relu -> d_out (fp32)
// ---------------------------------------------------------------------------
__global__ void apply_bn2(const float* __restrict__ out_raw,
                          const float* __restrict__ scale,
                          const float* __restrict__ shift,
                          float* __restrict__ out, long long total)
{
    long long gid = ((long long)blockIdx.x * 256 + threadIdx.x) * 4;
    if (gid >= total) return;
    int c0 = (int)(gid % 96);
    #pragma unroll
    for (int j = 0; j < 4; ++j) {
        float v = out_raw[gid + j] * scale[c0 + j] + shift[c0 + j];
        out[gid + j] = fmaxf(v, 0.f);
    }
}

// ---------------------------------------------------------------------------
static inline size_t al256(size_t x) { return (x + 255) & ~(size_t)255; }

extern "C" void kernel_launch(void* const* d_in, const int* in_sizes, int n_in,
                              void* d_out, int out_size, void* d_ws, size_t ws_size,
                              hipStream_t stream)
{
    const float* feats_a  = (const float*)d_in[0];
    const float* feats_b  = (const float*)d_in[1];
    const float* W_t      = (const float*)d_in[2];
    const float* bn1_g    = (const float*)d_in[3];
    const float* bn1_b    = (const float*)d_in[4];
    const float* W_out    = (const float*)d_in[5];
    const float* bn2_g    = (const float*)d_in[6];
    const float* bn2_b    = (const float*)d_in[7];
    const int*   a_idx    = (const int*)d_in[8];
    const int*   k_id     = (const int*)d_in[9];
    const int*   nbr_idx  = (const int*)d_in[10];
    const float* nbr_mask = (const float*)d_in[11];

    const int N = in_sizes[8];            // 600000
    const int ntiles = (N + 15) / 16;

    // ---- workspace layout ----
    char* ws = (char*)d_ws;
    size_t o = 0;
    __bf16* cat = (__bf16*)(ws + o);               o += al256((size_t)N * 128 * 2);
    float* up_raw  = (float*)(ws + o);             // aliased: disjoint lifetimes
    float* out_raw = (float*)(ws + o);             o += al256((size_t)N * 96 * 4);
    __bf16* WtP = (__bf16*)(ws + o);               o += al256((size_t)8  * 4 * 4 * 512 * 2);
    __bf16* WoP = (__bf16*)(ws + o);               o += al256((size_t)27 * 6 * 4 * 512 * 2);
    float* slots1 = (float*)(ws + o);  size_t s1b = al256((size_t)2 * 64 * NSLOT * 4); o += s1b;
    float* slots2 = (float*)(ws + o);  size_t s2b = al256((size_t)2 * 96 * NSLOT * 4); o += s2b;
    float* bn1_scale = (float*)(ws + o);           o += 256;
    float* bn1_shift = (float*)(ws + o);           o += 256;
    float* bn2_scale = (float*)(ws + o);           o += 256;
    float* bn2_shift = (float*)(ws + o);           o += 256;

    // zero BN-statistics slots every launch (graph-capturable memset node)
    hipMemsetAsync(slots1, 0, s1b + s2b, stream);

    // 1. pack weights into WMMA B-fragment order
    {
        int totT = 8 * 4 * 4 * 512;
        int totO = 27 * 6 * 4 * 512;
        pack_w<<<(totT + 255) / 256, 256, 0, stream>>>(W_t,   WtP, 128, 64, totT);
        pack_w<<<(totO + 255) / 256, 256, 0, stream>>>(W_out, WoP, 128, 96, totO);
    }

    // 2. stage 1 GEMM (8 masked weight banks) + BN1 stats
    {
        const int TPB = 8;
        int blocks = (ntiles + TPB - 1) / TPB;
        stage1_wmma<<<blocks, 128, 0, stream>>>(feats_a, a_idx, k_id, WtP,
                                                up_raw, slots1, N, ntiles, TPB);
    }

    // 3. BN1 finalize
    bn_finalize<<<1, 64, 0, stream>>>(slots1, 64, 1.0f / (float)N,
                                      bn1_g, bn1_b, bn1_scale, bn1_shift);

    // 4. build bf16 cat table (L2-resident)
    {
        long long tot = (long long)N * 16;
        int blocks = (int)((tot + 255) / 256);
        build_cat<<<blocks, 256, 0, stream>>>(up_raw, feats_b, bn1_scale, bn1_shift,
                                              cat, N);
    }

    // 5. stage 3: 27-tap gather GEMM (async-pipelined) + BN2 stats
    {
        const int TPB = 4;
        int blocks = (ntiles + TPB - 1) / TPB;
        stage3_wmma<<<blocks, 192, 0, stream>>>(cat, nbr_idx, nbr_mask, WoP,
                                                out_raw, slots2, N, ntiles, TPB);
    }

    // 6. BN2 finalize + apply
    bn_finalize<<<1, 96, 0, stream>>>(slots2, 96, 1.0f / (float)N,
                                      bn2_g, bn2_b, bn2_scale, bn2_shift);
    {
        long long tot = (long long)N * 96;
        int blocks = (int)((tot / 4 + 255) / 256);
        apply_bn2<<<blocks, 256, 0, stream>>>(out_raw, bn2_scale, bn2_shift,
                                              (float*)d_out, tot);
    }
}
